// StructureModule_36395552866969
// MI455X (gfx1250) — matrix operand order repaired and checked by reading, not compiled, not run on hardware
//
#include <hip/hip_runtime.h>

// ============================================================================
// AlphaFold-style Structure Module for MI455X (gfx1250, wave32, WMMA + TDM).
//  * pair tensor LayerNorm'ed ONCE -> bf16 cache (halves dominant HBM traffic)
//  * pair_bias GEMM hoisted out of the layer loop (z is loop-invariant)
//  * v_wmma_f32_16x16x32_bf16 GEMMs, K-depth-64 staging (2 WMMA/stage)
//  * A fragments = 2x ds_load_b128; B fragments = 8x ds_load_b32 (k-pair
//    interleaved LDS layout matching the WMMA VGPR pairing)
//  * o_pair z-tiles staged by the Tensor Data Mover (tensor_load_to_lds +
//    s_wait_tensorcnt) when the builtin exists; vector loads otherwise
//  * branchless edge handling (clamped loads + selects, no exec branching)
// ============================================================================

typedef __bf16 bf16;
typedef __attribute__((ext_vector_type(16))) __bf16 v16bf;
typedef __attribute__((ext_vector_type(8)))  __bf16 v8bf;
typedef __attribute__((ext_vector_type(8)))  float  v8f;
typedef __attribute__((ext_vector_type(4)))  unsigned int v4u;
typedef __attribute__((ext_vector_type(8)))  int i32x8;
typedef __attribute__((ext_vector_type(4)))  int i32x4;

#if defined(__has_builtin)
#if __has_builtin(__builtin_amdgcn_tensor_load_to_lds) && \
    __has_builtin(__builtin_amdgcn_s_wait_tensorcnt)
#define HAVE_TDM 1
#if defined(__has_include)
#if __has_include(<hip/amd_detail/amd_gfx1250_TDM.h>)
#define TDM_ARGS6 1
#endif
#endif
#endif
#endif

static constexpr int NRES = 768;
static constexpr int CSD  = 384;
static constexpr int CZD  = 128;
static constexpr int HH   = 12;
static constexpr int CC   = 16;
static constexpr int PQN  = 4;
static constexpr int PVN  = 8;
static constexpr int NLAY = 8;
static constexpr int LBINS= 50;
static constexpr int LCH  = 128;
static constexpr int FD   = 2112;              // H*(C+CZ+PV*4)
static constexpr int OF_O = 0, OF_OPL = 192, OF_ONORM = 480, OF_OPAIR = 576;
__device__ __host__ constexpr float WLC() { return 0.5773502691896258f; }   // sqrt(1/3)
__device__ __host__ constexpr float WCC() { return 0.23570226039551584f; }  // sqrt(2/(9*PQ))

static constexpr int ACC_FAPE0 = 0;   // ..7
static constexpr int ACC_CHL   = 8;
static constexpr int ACC_CONF  = 9;
static constexpr int ACC_PLDDT = 10;
static constexpr int ACC_N     = 16;

// ---------------------------------------------------------------- bf16 utils
__device__ inline bf16 f2bf(float x) {
  unsigned u = __float_as_uint(x);
  u = (u + 0x7FFFu + ((u >> 16) & 1u)) >> 16;
  unsigned short s = (unsigned short)u;
  bf16 r; __builtin_memcpy(&r, &s, 2);
  return r;
}
__device__ inline float bf2f(bf16 x) {
  unsigned short s; __builtin_memcpy(&s, &x, 2);
  return __uint_as_float(((unsigned)s) << 16);
}
__device__ inline unsigned bfbits(bf16 x) {
  unsigned short s; __builtin_memcpy(&s, &x, 2);
  return (unsigned)s;
}
// B-fragment K index (16-bit 32x16 B): half0 K 0..7,16..23, half1 K 8..15,24..31
__device__ inline int kmapB(int e, int half) { return (e & 7) + (half << 3) + ((e >> 3) << 4); }

__device__ inline v8f wmma_bf(v16bf a, v16bf b, v8f c) {
  return __builtin_amdgcn_wmma_f32_16x16x32_bf16(false, a, false, b, (short)0, c,
                                                 false, false);
}
__device__ inline v16bf cat16(v8bf lo, v8bf hi) {
  return __builtin_shufflevector(lo, hi, 0, 1, 2, 3, 4, 5, 6, 7,
                                 8, 9, 10, 11, 12, 13, 14, 15);
}
// Branchless guarded 8-element row load: clamp address, select zero.
__device__ inline v8bf load_row8(const bf16* __restrict__ P, long long rbase,
                                 int gc, int limit, bool rowok, bool fast) {
  v8bf val;
  if (fast && rowok && gc + 8 <= limit) {
    val = *(const v8bf*)(P + rbase + gc);
  } else {
#pragma unroll
    for (int e = 0; e < 8; e++) {
      int cc = gc + e;
      int cs = cc < limit ? cc : limit - 1;
      cs = cs < 0 ? 0 : cs;
      bf16 vv = P[rbase + cs];
      val[e] = (rowok && cc < limit) ? vv : f2bf(0.f);
    }
  }
  return val;
}

// ---------------------------------------------------------------- convert
__global__ void cvt_kernel(const float* __restrict__ x, bf16* __restrict__ y,
                           long long n) {
  long long i = (long long)blockIdx.x * blockDim.x + threadIdx.x;
  long long st = (long long)gridDim.x * blockDim.x;
  for (; i < n; i += st) y[i] = f2bf(x[i]);
}

// ---------------------------------------------------------------- layernorm
__global__ void ln_kernel(const float* __restrict__ x, const float* __restrict__ g,
                          const float* __restrict__ b, float* __restrict__ y,
                          bf16* __restrict__ yb, int cols) {
  long long row = blockIdx.x;
  const float* xr = x + row * cols;
  __shared__ float red[128];
  int tid = threadIdx.x;
  float s = 0.f;
  for (int c = tid; c < cols; c += 128) s += xr[c];
  red[tid] = s; __syncthreads();
  for (int o = 64; o > 0; o >>= 1) { if (tid < o) red[tid] += red[tid + o]; __syncthreads(); }
  float mu = red[0] / cols;
  __syncthreads();
  float v = 0.f;
  for (int c = tid; c < cols; c += 128) { float d = xr[c] - mu; v += d * d; }
  red[tid] = v; __syncthreads();
  for (int o = 64; o > 0; o >>= 1) { if (tid < o) red[tid] += red[tid + o]; __syncthreads(); }
  float rstd = rsqrtf(red[0] / cols + 1e-5f);
  for (int c = tid; c < cols; c += 128) {
    float o = (xr[c] - mu) * rstd * g[c] + b[c];
    if (y)  y[row * cols + c] = o;
    if (yb) yb[row * cols + c] = f2bf(o);
  }
}

// ---------------------------------------------------------------- GEMM (NN)
// C[M x Nn] = A[M x K] * B[K x Nn] (+bias)(+resid)(relu), bf16 in / f32 acc.
// 256 thr / 8 waves, 64x32 tile, K-depth 64 (2 WMMA per stage).
// B tile stored as interleaved k-pairs: u32 = {B[2k],B[2k+1]} per column,
// so a B fragment is 8x ds_load_b32 matching the WMMA VGPR pairing.
__global__ __launch_bounds__(256)
void gemm_nn(const bf16* __restrict__ A, int lda, long long sA,
             const bf16* __restrict__ B, int ldb, long long sB,
             const float* __restrict__ bias,
             const float* __restrict__ resid, int ldr,
             float* __restrict__ C, int ldc, long long sC,
             bf16* __restrict__ Cb, int ldcb, long long sCb,
             int M, int Nn, int K, int relu) {
  __shared__ alignas(16) bf16 As[64][72];
  __shared__ alignas(16) unsigned Bp[32][36];   // [k-pair][n] interleaved
  int z = blockIdx.z;
  A += (long long)z * sA;  B += (long long)z * sB;
  if (C)  C  += (long long)z * sC;
  if (Cb) Cb += (long long)z * sCb;
  int bm = blockIdx.y * 64, bn = blockIdx.x * 32;
  int tid = threadIdx.x, lane = tid & 31, wave = tid >> 5;
  int tr = wave >> 1, tc = wave & 1;
  int half = lane >> 4, lr = lane & 15;
  const bool fastA = ((lda & 7) == 0);
  const bool fastB = ((ldb & 7) == 0) && (bn + 32 <= Nn);
  v8f acc = {};
  for (int k0 = 0; k0 < K; k0 += 64) {
    // A tile 64x64
#pragma unroll
    for (int it = 0; it < 2; it++) {
      int idx = tid + it * 256;
      int r = idx >> 3, cg = (idx & 7) << 3;
      int gr = bm + r;
      int grc = gr < M ? gr : M - 1;
      v8bf val = load_row8(A, (long long)grc * lda, k0 + cg, K, gr < M, fastA);
      *(v8bf*)&As[r][cg] = val;
    }
    // B tile 64x32 -> interleaved pairs
    if (tid < 128) {
      int kp = tid >> 2, ng = (tid & 3) << 3;
      int gk = k0 + kp * 2, gn = bn + ng;
      v8bf vlo, vhi;
      if (fastB && gk + 2 <= K) {
        vlo = *(const v8bf*)(B + (long long)gk * ldb + gn);
        vhi = *(const v8bf*)(B + (long long)(gk + 1) * ldb + gn);
      } else {
        int g0c = gk < K ? gk : K - 1;
        int g1 = gk + 1;
        int g1c = g1 < K ? g1 : K - 1;
#pragma unroll
        for (int e = 0; e < 8; e++) {
          int cn = gn + e;
          int cns = cn < Nn ? cn : Nn - 1;
          cns = cns < 0 ? 0 : cns;
          bf16 a0 = B[(long long)g0c * ldb + cns];
          bf16 a1 = B[(long long)g1c * ldb + cns];
          vlo[e] = (gk < K && cn < Nn) ? a0 : f2bf(0.f);
          vhi[e] = (g1 < K && cn < Nn) ? a1 : f2bf(0.f);
        }
      }
      v4u p0, p1;
#pragma unroll
      for (int e = 0; e < 4; e++) p0[e] = bfbits(vlo[e]) | (bfbits(vhi[e]) << 16);
#pragma unroll
      for (int e = 0; e < 4; e++) p1[e] = bfbits(vlo[e + 4]) | (bfbits(vhi[e + 4]) << 16);
      *(v4u*)&Bp[kp][ng] = p0;
      *(v4u*)&Bp[kp][ng + 4] = p1;
    }
    if (k0 + 64 < K)
      __builtin_prefetch(A + (long long)(bm + (tid & 63)) * lda + k0 + 64, 0, 3);
    __syncthreads();
    int row = tr * 16 + lr, col = tc * 16 + lr;
#pragma unroll
    for (int kk = 0; kk < 2; kk++) {
      v8bf alo = *(const v8bf*)&As[row][kk * 32 + half * 8];
      v8bf ahi = *(const v8bf*)&As[row][kk * 32 + 16 + half * 8];
      v16bf af = cat16(alo, ahi);
      union { unsigned u[8]; v16bf v; } bu;
#pragma unroll
      for (int m = 0; m < 8; m++) bu.u[m] = Bp[kk * 16 + half * 4 + m + (m & 4)][col];
      acc = wmma_bf(af, bu.v, acc);
    }
    __syncthreads();
  }
#pragma unroll
  for (int j = 0; j < 8; j++) {
    int gr = bm + tr * 16 + j + 8 * half;
    int gc = bn + tc * 16 + lr;
    if (gr < M && gc < Nn) {
      float v = acc[j];
      if (bias)  v += bias[gc];
      if (resid) v += resid[(long long)gr * ldr + gc];
      if (relu)  v = v > 0.f ? v : 0.f;
      if (C)  C[(long long)gr * ldc + gc] = v;
      if (Cb) Cb[(long long)gr * ldcb + gc] = f2bf(v);
    }
  }
}

// ---------------------------------------------------------------- GEMM (NT)
// C[M x Nn] = A[M x K] * B^T, B stored [Nn x K]. Used for logits (K = 32).
__global__ __launch_bounds__(256)
void gemm_nt(const bf16* __restrict__ A, int lda, long long sA,
             const bf16* __restrict__ B, int ldb, long long sB,
             float* __restrict__ C, int ldc, long long sC,
             int M, int Nn, int K) {
  __shared__ alignas(16) bf16 As[64][40];
  __shared__ alignas(16) bf16 Bs[32][40];   // [n][k] -> fragment is contiguous
  int z = blockIdx.z;
  A += (long long)z * sA;  B += (long long)z * sB;  C += (long long)z * sC;
  int bm = blockIdx.y * 64, bn = blockIdx.x * 32;
  int tid = threadIdx.x, lane = tid & 31, wave = tid >> 5;
  int tr = wave >> 1, tc = wave & 1;
  int half = lane >> 4, lr = lane & 15;
  const bool fastA = ((lda & 7) == 0);
  const bool fastB = ((ldb & 7) == 0);
  v8f acc = {};
  for (int k0 = 0; k0 < K; k0 += 32) {
    {
      int r = tid >> 2, cg = (tid & 3) << 3;
      int gr = bm + r;
      int grc = gr < M ? gr : M - 1;
      v8bf val = load_row8(A, (long long)grc * lda, k0 + cg, K, gr < M, fastA);
      *(v8bf*)&As[r][cg] = val;
    }
    if (tid < 128) {
      int n = tid >> 2, kg = (tid & 3) << 3;
      int gn = bn + n;
      int gnc = gn < Nn ? gn : Nn - 1;
      v8bf val = load_row8(B, (long long)gnc * ldb, k0 + kg, K, gn < Nn, fastB);
      *(v8bf*)&Bs[n][kg] = val;
    }
    __syncthreads();
    int row = tr * 16 + lr, col = tc * 16 + lr;
    v8bf alo = *(const v8bf*)&As[row][half * 16];
    v8bf ahi = *(const v8bf*)&As[row][half * 16 + 8];
    v16bf af = cat16(alo, ahi);
    v16bf bfr = *(const v16bf*)&Bs[col][half * 16];
    acc = wmma_bf(af, bfr, acc);
    __syncthreads();
  }
#pragma unroll
  for (int j = 0; j < 8; j++) {
    int gr = bm + tr * 16 + j + 8 * half;
    int gc = bn + tc * 16 + lr;
    if (gr < M && gc < Nn) C[(long long)gr * ldc + gc] = acc[j];
  }
}

// ---------------------------------------------------------------- o_pair
// o_pair[i] (H x CZ) = a[:, i, :] (H x N) @ z_ln[i] (N x CZ); BM=16, BN=128.
// The 64x128 bf16 z-tile is staged by the Tensor Data Mover when available.
__global__ __launch_bounds__(256)
void opair_kernel(const bf16* __restrict__ a,    // [H][N][N]
                  const bf16* __restrict__ zb,   // [N*N][CZ]
                  bf16* __restrict__ feats) {    // [N][FD]
  int i = blockIdx.x;
  __shared__ alignas(16) bf16 As[16][72];
  __shared__ alignas(16) bf16 Bs[64][CZD];
  int tid = threadIdx.x, lane = tid & 31, wave = tid >> 5;
  int half = lane >> 4, lr = lane & 15;
  const long long NN = (long long)NRES * NRES;
  v8f acc = {};
  for (int k0 = 0; k0 < NRES; k0 += 64) {
    if (tid < 128) {
      int r = tid >> 3, cg = (tid & 7) << 3;
      int rc = r < HH ? r : HH - 1;
      v8bf val = *(const v8bf*)(a + (long long)rc * NN + (long long)i * NRES + k0 + cg);
      if (r >= HH) {
#pragma unroll
        for (int e = 0; e < 8; e++) val[e] = f2bf(0.f);
      }
      *(v8bf*)&As[r][cg] = val;
    }
#ifdef HAVE_TDM
    if (wave == 0) {
      // Tensor DMA: 2-D tile 128 (elems) x 64 (rows), 2-byte elements.
      unsigned long long ga =
          (unsigned long long)(zb + ((long long)i * NRES + k0) * CZD);
      unsigned ldsa = (unsigned)(unsigned long long)
          (__attribute__((address_space(3))) bf16*)&Bs[0][0];
      v4u g0;
      g0[0] = 1u;                                   // count=1, user descriptor
      g0[1] = ldsa;                                 // lds_addr
      g0[2] = (unsigned)ga;                         // global_addr[31:0]
      g0[3] = (unsigned)((ga >> 32) & 0x01FFFFFFull) | 0x80000000u; // [56:32]|type=2
      unsigned long long q0 = 0x10000ull | (128ull << 48);  // data_size=2B, tdim0=128
      unsigned long long q1 = (64ull << 16) | (128ull << 48); // tdim1=64, tile0=128
      unsigned long long q2 = 64ull | (128ull << 32);         // tile1=64, stride0=128
      i32x8 g1;
      g1[0] = (int)q0; g1[1] = (int)(q0 >> 32);
      g1[2] = (int)q1; g1[3] = (int)(q1 >> 32);
      g1[4] = (int)q2; g1[5] = (int)(q2 >> 32);
      g1[6] = 0;       g1[7] = 0;
      i32x4 z4 = {0, 0, 0, 0};
#ifdef TDM_ARGS6
      i32x8 z8 = {0, 0, 0, 0, 0, 0, 0, 0};
      __builtin_amdgcn_tensor_load_to_lds(g0, g1, z4, z4, z8, 0);
#else
      __builtin_amdgcn_tensor_load_to_lds(g0, g1, z4, z4, 0);
#endif
      __builtin_amdgcn_s_wait_tensorcnt(0);
    }
    asm volatile("" ::: "memory");   // LDS written by TDM, not visible to LLVM
#else
#pragma unroll
    for (int it = 0; it < 4; it++) {
      int idx = tid + it * 256;
      int r = idx >> 4, cg = (idx & 15) << 3;
      *(v8bf*)&Bs[r][cg] =
          *(const v8bf*)(zb + ((long long)i * NRES + k0 + r) * CZD + cg);
    }
#endif
    __syncthreads();
#pragma unroll
    for (int kk = 0; kk < 2; kk++) {
      v8bf alo = *(const v8bf*)&As[lr][kk * 32 + half * 8];
      v8bf ahi = *(const v8bf*)&As[lr][kk * 32 + 16 + half * 8];
      v16bf af = cat16(alo, ahi);
      v16bf bfr;
#pragma unroll
      for (int e = 0; e < 16; e++)
        bfr[e] = Bs[kk * 32 + kmapB(e, half)][wave * 16 + lr];
      acc = wmma_bf(af, bfr, acc);
    }
    __syncthreads();
  }
#pragma unroll
  for (int j = 0; j < 8; j++) {
    int h = j + 8 * half;
    int c = wave * 16 + lr;
    if (h < HH)
      feats[(long long)i * FD + OF_OPAIR + h * CZD + c] = f2bf(acc[j]);
  }
}

// ---------------------------------------------------------------- points
__global__ void points_kernel(const float* __restrict__ q, const float* __restrict__ k,
                              const float* __restrict__ qp, const float* __restrict__ kp,
                              const float* __restrict__ vp,
                              const float* __restrict__ R, const float* __restrict__ t,
                              const float* __restrict__ gamma_p,
                              bf16* __restrict__ U, bf16* __restrict__ V,
                              bf16* __restrict__ vgb,
                              float* __restrict__ qn, float* __restrict__ kn) {
  int i = blockIdx.x, tid = threadIdx.x;
  __shared__ float Rs[9], ts[3], sq[HH], sk[HH], gam[HH];
  if (tid < 9) Rs[tid] = R[i * 9 + tid];
  if (tid < 3) ts[tid] = t[i * 3 + tid];
  if (tid < HH) { sq[tid] = 0.f; sk[tid] = 0.f; gam[tid] = log1pf(expf(gamma_p[tid])); }
  __syncthreads();
  if (tid < HH * PQN) {                 // 48 query/key points
    int h = tid >> 2;
    const float* qq = qp + (long long)i * (HH * PQN * 3) + tid * 3;
    float x = qq[0], y = qq[1], zc = qq[2];
    float gx = Rs[0] * x + Rs[1] * y + Rs[2] * zc + ts[0];
    float gy = Rs[3] * x + Rs[4] * y + Rs[5] * zc + ts[1];
    float gz = Rs[6] * x + Rs[7] * y + Rs[8] * zc + ts[2];
    atomicAdd(&sq[h], gx * gx + gy * gy + gz * gz);
    float sc = WLC() * WCC() * gam[h];
    bf16* Uh = U + ((long long)h * NRES + i) * 32;
    int p = tid & 3;
    Uh[16 + p * 3 + 0] = f2bf(gx * sc);
    Uh[16 + p * 3 + 1] = f2bf(gy * sc);
    Uh[16 + p * 3 + 2] = f2bf(gz * sc);
    const float* kk = kp + (long long)i * (HH * PQN * 3) + tid * 3;
    x = kk[0]; y = kk[1]; zc = kk[2];
    gx = Rs[0] * x + Rs[1] * y + Rs[2] * zc + ts[0];
    gy = Rs[3] * x + Rs[4] * y + Rs[5] * zc + ts[1];
    gz = Rs[6] * x + Rs[7] * y + Rs[8] * zc + ts[2];
    atomicAdd(&sk[h], gx * gx + gy * gy + gz * gz);
    bf16* Vh = V + ((long long)h * NRES + i) * 32;
    Vh[16 + p * 3 + 0] = f2bf(gx);
    Vh[16 + p * 3 + 1] = f2bf(gy);
    Vh[16 + p * 3 + 2] = f2bf(gz);
  }
  for (int c = tid; c < HH * CC; c += blockDim.x) {
    int h = c >> 4, cc = c & 15;
    U[((long long)h * NRES + i) * 32 + cc] = f2bf(q[(long long)i * (HH * CC) + c] * (WLC() * 0.25f));
    V[((long long)h * NRES + i) * 32 + cc] = f2bf(k[(long long)i * (HH * CC) + c]);
  }
  for (int c = tid; c < HH * 4; c += blockDim.x) {
    int h = c >> 2, cc = 28 + (c & 3);
    U[((long long)h * NRES + i) * 32 + cc] = f2bf(0.f);
    V[((long long)h * NRES + i) * 32 + cc] = f2bf(0.f);
  }
  if (tid < HH * PVN) {                 // 96 value points
    const float* vv = vp + (long long)i * (HH * PVN * 3) + tid * 3;
    float x = vv[0], y = vv[1], zc = vv[2];
    float gx = Rs[0] * x + Rs[1] * y + Rs[2] * zc + ts[0];
    float gy = Rs[3] * x + Rs[4] * y + Rs[5] * zc + ts[1];
    float gz = Rs[6] * x + Rs[7] * y + Rs[8] * zc + ts[2];
    bf16* o = vgb + (long long)i * (HH * PVN * 3) + tid * 3;
    o[0] = f2bf(gx); o[1] = f2bf(gy); o[2] = f2bf(gz);
  }
  __syncthreads();
  if (tid < HH) {
    qn[(long long)tid * NRES + i] = sq[tid];
    kn[(long long)tid * NRES + i] = sk[tid];
  }
}

// ---------------------------------------------------------------- softmax
__global__ __launch_bounds__(256)
void softmax_kernel(const float* __restrict__ dot, const bf16* __restrict__ biasz,
                    const float* __restrict__ qn, const float* __restrict__ kn,
                    const float* __restrict__ gamma_p, bf16* __restrict__ a) {
  int h = blockIdx.x / NRES, i = blockIdx.x % NRES;
  int tid = threadIdx.x;
  __shared__ float lrow[NRES];
  __shared__ float red[256];
  const float* drow = dot + ((long long)h * NRES + i) * NRES;
  float gam = log1pf(expf(gamma_p[h]));
  float ct = 0.5f * WLC() * WCC() * gam;
  float qni = qn[(long long)h * NRES + i];
  float lm = -1e30f;
  for (int j = tid; j < NRES; j += 256) {
    float l = drow[j] + WLC() * bf2f(biasz[((long long)i * NRES + j) * HH + h])
            - ct * (qni + kn[(long long)h * NRES + j]);
    lrow[j] = l;
    lm = fmaxf(lm, l);
  }
  red[tid] = lm; __syncthreads();
  for (int o = 128; o > 0; o >>= 1) { if (tid < o) red[tid] = fmaxf(red[tid], red[tid + o]); __syncthreads(); }
  float mx = red[0]; __syncthreads();
  float se = 0.f;
  for (int j = tid; j < NRES; j += 256) { float e = expf(lrow[j] - mx); lrow[j] = e; se += e; }
  red[tid] = se; __syncthreads();
  for (int o = 128; o > 0; o >>= 1) { if (tid < o) red[tid] += red[tid + o]; __syncthreads(); }
  float inv = 1.f / red[0];
  bf16* arow = a + ((long long)h * NRES + i) * NRES;
  for (int j = tid; j < NRES; j += 256) arow[j] = f2bf(lrow[j] * inv);
}

// ---------------------------------------------------------------- feats asm
__global__ void feats_kernel(const float* __restrict__ o, const float* __restrict__ opg,
                             const float* __restrict__ R, const float* __restrict__ t,
                             bf16* __restrict__ feats) {
  int i = blockIdx.x, tid = threadIdx.x;
  __shared__ float Rs[9], ts[3];
  if (tid < 9) Rs[tid] = R[i * 9 + tid];
  if (tid < 3) ts[tid] = t[i * 3 + tid];
  __syncthreads();
  if (tid < HH * CC)
    feats[(long long)i * FD + OF_O + tid] = f2bf(o[(long long)i * (HH * CC) + tid]);
  if (tid < HH * PVN) {
    const float* g = opg + (long long)i * (HH * PVN * 3) + tid * 3;
    float dx = g[0] - ts[0], dy = g[1] - ts[1], dz = g[2] - ts[2];
    float lx = Rs[0] * dx + Rs[3] * dy + Rs[6] * dz;
    float ly = Rs[1] * dx + Rs[4] * dy + Rs[7] * dz;
    float lz = Rs[2] * dx + Rs[5] * dy + Rs[8] * dz;
    bf16* f = feats + (long long)i * FD;
    f[OF_OPL + tid * 3 + 0] = f2bf(lx);
    f[OF_OPL + tid * 3 + 1] = f2bf(ly);
    f[OF_OPL + tid * 3 + 2] = f2bf(lz);
    f[OF_ONORM + tid] = f2bf(sqrtf(lx * lx + ly * ly + lz * lz + 1e-8f));
  }
}

// ---------------------------------------------------------------- backbone
__global__ void backbone_kernel(const float* __restrict__ u, float* __restrict__ R,
                                float* __restrict__ t) {
  int i = blockIdx.x * 64 + threadIdx.x;
  if (i >= NRES) return;
  float b = u[i * 6 + 0], c = u[i * 6 + 1], d = u[i * 6 + 2];
  float tux = u[i * 6 + 3], tuy = u[i * 6 + 4], tuz = u[i * 6 + 5];
  float inv = rsqrtf(1.f + b * b + c * c + d * d);
  float a = inv; b *= inv; c *= inv; d *= inv;
  float Ru[9] = { a*a + b*b - c*c - d*d, 2.f*(b*c - a*d),       2.f*(b*d + a*c),
                  2.f*(b*c + a*d),       a*a - b*b + c*c - d*d, 2.f*(c*d - a*b),
                  2.f*(b*d - a*c),       2.f*(c*d + a*b),       a*a - b*b - c*c + d*d };
  float Ro[9];
#pragma unroll
  for (int k = 0; k < 9; k++) Ro[k] = R[i * 9 + k];
  float Rn[9];
#pragma unroll
  for (int x = 0; x < 3; x++)
#pragma unroll
    for (int zc = 0; zc < 3; zc++)
      Rn[x * 3 + zc] = Ro[x * 3 + 0] * Ru[0 * 3 + zc] + Ro[x * 3 + 1] * Ru[1 * 3 + zc] +
                       Ro[x * 3 + 2] * Ru[2 * 3 + zc];
  float tn[3];
#pragma unroll
  for (int x = 0; x < 3; x++)
    tn[x] = Ro[x * 3 + 0] * tux + Ro[x * 3 + 1] * tuy + Ro[x * 3 + 2] * tuz + t[i * 3 + x];
#pragma unroll
  for (int k = 0; k < 9; k++) R[i * 9 + k] = Rn[k];
#pragma unroll
  for (int k = 0; k < 3; k++) t[i * 3 + k] = tn[k];
}

// ---------------------------------------------------------------- FAPE
__global__ __launch_bounds__(256)
void fape_kernel(const float* __restrict__ R, const float* __restrict__ t,
                 const float* __restrict__ Rt, const float* __restrict__ tt,
                 float* __restrict__ acc) {
  int i = blockIdx.x, tid = threadIdx.x;
  __shared__ float red[256];
  __shared__ float Ri[9], Qi[9], ti3[3], si3[3];
  if (tid < 9) { Ri[tid] = R[i * 9 + tid]; Qi[tid] = Rt[i * 9 + tid]; }
  if (tid < 3) { ti3[tid] = t[i * 3 + tid]; si3[tid] = tt[i * 3 + tid]; }
  __syncthreads();
  float s = 0.f;
  for (int j = tid; j < NRES; j += 256) {
    float dx = t[j * 3 + 0] - ti3[0], dy = t[j * 3 + 1] - ti3[1], dz = t[j * 3 + 2] - ti3[2];
    float px = Ri[0] * dx + Ri[3] * dy + Ri[6] * dz;
    float py = Ri[1] * dx + Ri[4] * dy + Ri[7] * dz;
    float pz = Ri[2] * dx + Ri[5] * dy + Ri[8] * dz;
    float ex = tt[j * 3 + 0] - si3[0], ey = tt[j * 3 + 1] - si3[1], ez = tt[j * 3 + 2] - si3[2];
    float qx = Qi[0] * ex + Qi[3] * ey + Qi[6] * ez;
    float qy = Qi[1] * ex + Qi[4] * ey + Qi[7] * ez;
    float qz = Qi[2] * ex + Qi[5] * ey + Qi[8] * ez;
    float ax = px - qx, ay = py - qy, az = pz - qz;
    float d = sqrtf(ax * ax + ay * ay + az * az + 1e-12f);
    s += fminf(d, 10.f);
  }
  red[tid] = s; __syncthreads();
  for (int o = 128; o > 0; o >>= 1) { if (tid < o) red[tid] += red[tid + o]; __syncthreads(); }
  if (tid == 0) atomicAdd(acc, red[0]);
}

// ---------------------------------------------------------------- lddt true
__global__ __launch_bounds__(256)
void lddt_true_kernel(const float* __restrict__ t, const float* __restrict__ tt,
                      float* __restrict__ ltrue, float* __restrict__ accs) {
  int i = blockIdx.x, tid = threadIdx.x;
  __shared__ float rs[256], rm[256];
  float tx = t[i * 3], ty = t[i * 3 + 1], tz = t[i * 3 + 2];
  float sx = tt[i * 3], sy = tt[i * 3 + 1], sz = tt[i * 3 + 2];
  float ssum = 0.f, msum = 0.f;
  for (int j = tid; j < NRES; j += 256) {
    float ax = tx - t[j * 3], ay = ty - t[j * 3 + 1], az = tz - t[j * 3 + 2];
    float dp = sqrtf(ax * ax + ay * ay + az * az + 1e-10f);
    float bx = sx - tt[j * 3], by = sy - tt[j * 3 + 1], bz = sz - tt[j * 3 + 2];
    float dt = sqrtf(bx * bx + by * by + bz * bz + 1e-10f);
    if (dt < 15.f && j != i) {
      float diff = fabsf(dp - dt);
      float sc = ((diff < 0.5f ? 1.f : 0.f) + (diff < 1.f ? 1.f : 0.f) +
                  (diff < 2.f ? 1.f : 0.f) + (diff < 4.f ? 1.f : 0.f)) * 0.25f;
      ssum += sc; msum += 1.f;
    }
  }
  rs[tid] = ssum; rm[tid] = msum; __syncthreads();
  for (int o = 128; o > 0; o >>= 1) {
    if (tid < o) { rs[tid] += rs[tid + o]; rm[tid] += rm[tid + o]; }
    __syncthreads();
  }
  if (tid == 0) {
    float v = rs[0] / (rm[0] + 1e-10f);
    ltrue[i] = v;
    atomicAdd(&accs[ACC_CHL], v);
  }
}

// ---------------------------------------------------------------- confidence
__global__ void conf_kernel(const float* __restrict__ lgt, const float* __restrict__ ltrue,
                            float* __restrict__ accs) {
  int i = blockIdx.x, tid = threadIdx.x;   // 64 threads
  const float* l = lgt + (long long)i * LBINS;
  __shared__ float red[64];
  float m = (tid < LBINS) ? l[tid] : -1e30f;
  red[tid] = m; __syncthreads();
  for (int o = 32; o > 0; o >>= 1) { if (tid < o) red[tid] = fmaxf(red[tid], red[tid + o]); __syncthreads(); }
  float mx = red[0]; __syncthreads();
  float e = (tid < LBINS) ? expf(l[tid] - mx) : 0.f;
  red[tid] = e; __syncthreads();
  for (int o = 32; o > 0; o >>= 1) { if (tid < o) red[tid] += red[tid + o]; __syncthreads(); }
  float Z = red[0]; __syncthreads();
  float w = (tid < LBINS) ? (e / Z) * ((tid + 0.5f) / LBINS) : 0.f;
  red[tid] = w; __syncthreads();
  for (int o = 32; o > 0; o >>= 1) { if (tid < o) red[tid] += red[tid + o]; __syncthreads(); }
  if (tid == 0) {
    int bin = (int)(ltrue[i] * LBINS);
    bin = bin < 0 ? 0 : (bin > LBINS - 1 ? LBINS - 1 : bin);
    float logp = l[bin] - mx - logf(Z);
    atomicAdd(&accs[ACC_CONF], -logp);
    atomicAdd(&accs[ACC_PLDDT], red[0]);
  }
}

// ---------------------------------------------------------------- misc
__global__ void init_kernel(float* __restrict__ R, float* __restrict__ t,
                            float* __restrict__ accs) {
  int i = blockIdx.x * 64 + threadIdx.x;
  if (blockIdx.x == 0 && threadIdx.x < ACC_N) accs[threadIdx.x] = 0.f;
  if (i < NRES) {
#pragma unroll
    for (int k = 0; k < 9; k++) R[i * 9 + k] = (k == 0 || k == 4 || k == 8) ? 1.f : 0.f;
#pragma unroll
    for (int k = 0; k < 3; k++) t[i * 3 + k] = 0.f;
  }
}

__global__ void coords_kernel(const float* __restrict__ R, const float* __restrict__ t,
                              const float* __restrict__ local, float* __restrict__ out) {
  int i = blockIdx.x, tid = threadIdx.x;
  if (tid < 9) {
    int a = tid / 3, x = tid % 3;
    const float* Ri = R + i * 9;
    const float* L = local + (long long)i * 9;
    float v = Ri[x * 3 + 0] * L[a * 3 + 0] + Ri[x * 3 + 1] * L[a * 3 + 1] +
              Ri[x * 3 + 2] * L[a * 3 + 2] + t[i * 3 + x];
    out[(long long)i * 9 + tid] = v;
  }
}

__global__ void finalize_kernel(const float* __restrict__ accs, float* __restrict__ out) {
  if (blockIdx.x == 0 && threadIdx.x == 0) {
    float inv = 1.f / ((float)NRES * (float)NRES) / 10.f;
    float aux = 0.f;
    for (int l = 0; l < NLAY; l++) aux += accs[ACC_FAPE0 + l] * inv;
    aux /= (float)NLAY;
    out[6912] = accs[ACC_PLDDT] / (float)NRES;      // chain_plddt
    out[6913] = accs[ACC_CHL] / (float)NRES;        // chain_lddt
    out[6914] = accs[ACC_FAPE0 + NLAY - 1] * inv;   // fape
    out[6915] = accs[ACC_CONF] / (float)NRES;       // conf_loss
    out[6916] = aux;                                // aux_loss
  }
}

// ============================================================================
// Host orchestration (JAX pytree flatten, sorted dict keys)
// ============================================================================
enum {
  IN_LOCAL = 0, IN_PAIR = 1, IN_BB_B = 2, IN_BB_W = 3, IN_GAMMA = 4,
  IN_K_B = 5, IN_K_W = 6, IN_KP_B = 7, IN_KP_W = 8, IN_OUT_B = 9, IN_OUT_W = 10,
  IN_PB_B = 11, IN_PB_W = 12, IN_Q_B = 13, IN_Q_W = 14, IN_QP_B = 15, IN_QP_W = 16,
  IN_V_B = 17, IN_V_W = 18, IN_VP_B = 19, IN_VP_W = 20,
  IN_L0_B = 21, IN_L0_W = 22, IN_L1_B = 23, IN_L1_W = 24, IN_L2_B = 25, IN_L2_W = 26,
  IN_LN1_B = 27, IN_LN1_G = 28, IN_LN2_B = 29, IN_LN2_G = 30,
  IN_LNP_B = 31, IN_LNP_G = 32, IN_LNS_B = 33, IN_LNS_G = 34,
  IN_SL_B = 35, IN_SL_W = 36,
  IN_T0_B = 37, IN_T0_W = 38, IN_T1_B = 39, IN_T1_W = 40, IN_T2_B = 41, IN_T2_W = 42,
  IN_ROT = 43, IN_SINGLE = 44, IN_TRANS = 45
};

static inline void launch_gemm(hipStream_t st, const bf16* A, int lda, long long sA,
                               const bf16* B, int ldb, long long sB,
                               const float* bias, const float* resid, int ldr,
                               float* C, int ldc, long long sC,
                               bf16* Cb, int ldcb, long long sCb,
                               int M, int Nn, int K, int relu, int batch) {
  dim3 grid((Nn + 31) / 32, (M + 63) / 64, batch);
  gemm_nn<<<grid, dim3(256), 0, st>>>(A, lda, sA, B, ldb, sB, bias, resid, ldr,
                                      C, ldc, sC, Cb, ldcb, sCb, M, Nn, K, relu);
}

static inline void launch_cvt(hipStream_t st, const float* x, bf16* y, long long n) {
  long long blocks = (n + 255) / 256;
  if (blocks > 4096) blocks = 4096;
  cvt_kernel<<<dim3((unsigned)blocks), dim3(256), 0, st>>>(x, y, n);
}

extern "C" void kernel_launch(void* const* d_in, const int* in_sizes, int n_in,
                              void* d_out, int out_size, void* d_ws, size_t ws_size,
                              hipStream_t stream) {
  (void)in_sizes; (void)n_in; (void)out_size; (void)ws_size;
  const float* F[48];
  for (int i = 0; i < 46; i++) F[i] = (const float*)d_in[i];
  float* out = (float*)d_out;

  size_t off = 0;
  auto alloc = [&](size_t bytes) {
    void* r = (char*)d_ws + off;
    off += (bytes + 255) & ~(size_t)255;
    return r;
  };
  const long long NN = (long long)NRES * NRES;
  bf16*  zbf    = (bf16*)alloc(NN * CZD * 2);
  bf16*  biaszb = (bf16*)alloc(NN * HH * 2);
  float* dotf   = (float*)alloc((long long)HH * NN * 4);
  bf16*  abf    = (bf16*)alloc((long long)HH * NN * 2);
  float* s      = (float*)alloc((long long)NRES * CSD * 4);
  bf16*  sbf    = (bf16*)alloc((long long)NRES * CSD * 2);
  float* tmp    = (float*)alloc((long long)NRES * CSD * 4);
  bf16*  tmpbf  = (bf16*)alloc((long long)NRES * CSD * 2);
  bf16* wsl  = (bf16*)alloc(384 * 384 * 2);
  bf16* wq   = (bf16*)alloc(384 * 192 * 2);
  bf16* wk   = (bf16*)alloc(384 * 192 * 2);
  bf16* wv   = (bf16*)alloc(384 * 192 * 2);
  bf16* wqp  = (bf16*)alloc(384 * 144 * 2);
  bf16* wkp  = (bf16*)alloc(384 * 144 * 2);
  bf16* wvp  = (bf16*)alloc(384 * 288 * 2);
  bf16* wpb  = (bf16*)alloc(128 * 12 * 2);
  bf16* wout = (bf16*)alloc((long long)FD * 384 * 2);
  bf16* wt0  = (bf16*)alloc(384 * 384 * 2);
  bf16* wt1  = (bf16*)alloc(384 * 384 * 2);
  bf16* wt2  = (bf16*)alloc(384 * 384 * 2);
  bf16* wbb  = (bf16*)alloc(384 * 6 * 2);
  bf16* wl0  = (bf16*)alloc(384 * 128 * 2);
  bf16* wl1  = (bf16*)alloc(128 * 128 * 2);
  bf16* wl2  = (bf16*)alloc(128 * 50 * 2);
  float* qf   = (float*)alloc((long long)NRES * 192 * 4);
  float* kf   = (float*)alloc((long long)NRES * 192 * 4);
  float* qpf  = (float*)alloc((long long)NRES * 144 * 4);
  float* kpf  = (float*)alloc((long long)NRES * 144 * 4);
  float* vpf  = (float*)alloc((long long)NRES * 288 * 4);
  bf16*  vbf  = (bf16*)alloc((long long)NRES * 192 * 2);
  bf16*  vgbf = (bf16*)alloc((long long)NRES * 288 * 2);
  bf16*  Ub   = (bf16*)alloc((long long)HH * NRES * 32 * 2);
  bf16*  Vb   = (bf16*)alloc((long long)HH * NRES * 32 * 2);
  float* qn   = (float*)alloc((long long)HH * NRES * 4);
  float* kn   = (float*)alloc((long long)HH * NRES * 4);
  float* of   = (float*)alloc((long long)NRES * 192 * 4);
  float* opgf = (float*)alloc((long long)NRES * 288 * 4);
  bf16*  featsb = (bf16*)alloc((long long)NRES * FD * 2);
  bf16*  h1b  = (bf16*)alloc((long long)NRES * CSD * 2);
  bf16*  h2b  = (bf16*)alloc((long long)NRES * CSD * 2);
  float* uf   = (float*)alloc((long long)NRES * 6 * 4);
  float* Rc   = (float*)alloc((long long)NRES * 9 * 4);
  float* tc   = (float*)alloc((long long)NRES * 3 * 4);
  bf16*  l1b  = (bf16*)alloc((long long)NRES * LCH * 2);
  bf16*  l2b  = (bf16*)alloc((long long)NRES * LCH * 2);
  float* lgt  = (float*)alloc((long long)NRES * LBINS * 4);
  float* ltru = (float*)alloc((long long)NRES * 4);
  float* accs = (float*)alloc(ACC_N * 4);

  launch_cvt(stream, F[IN_SL_W],  wsl,  384 * 384);
  launch_cvt(stream, F[IN_Q_W],   wq,   384 * 192);
  launch_cvt(stream, F[IN_K_W],   wk,   384 * 192);
  launch_cvt(stream, F[IN_V_W],   wv,   384 * 192);
  launch_cvt(stream, F[IN_QP_W],  wqp,  384 * 144);
  launch_cvt(stream, F[IN_KP_W],  wkp,  384 * 144);
  launch_cvt(stream, F[IN_VP_W],  wvp,  384 * 288);
  launch_cvt(stream, F[IN_PB_W],  wpb,  128 * 12);
  launch_cvt(stream, F[IN_OUT_W], wout, (long long)FD * 384);
  launch_cvt(stream, F[IN_T0_W],  wt0,  384 * 384);
  launch_cvt(stream, F[IN_T1_W],  wt1,  384 * 384);
  launch_cvt(stream, F[IN_T2_W],  wt2,  384 * 384);
  launch_cvt(stream, F[IN_BB_W],  wbb,  384 * 6);
  launch_cvt(stream, F[IN_L0_W],  wl0,  384 * 128);
  launch_cvt(stream, F[IN_L1_W],  wl1,  128 * 128);
  launch_cvt(stream, F[IN_L2_W],  wl2,  128 * 50);

  init_kernel<<<dim3(12), dim3(64), 0, stream>>>(Rc, tc, accs);

  // loop-invariant pair work (once, not per layer)
  ln_kernel<<<dim3((unsigned)NN), dim3(128), 0, stream>>>(
      F[IN_PAIR], F[IN_LNP_G], F[IN_LNP_B], nullptr, zbf, CZD);
  launch_gemm(stream, zbf, CZD, 0, wpb, 12, 0, F[IN_PB_B], nullptr, 0,
              nullptr, 0, 0, biaszb, HH, 0, (int)NN, HH, CZD, 0, 1);

  // s = LN(single) @ Wsl + b
  ln_kernel<<<dim3(NRES), dim3(128), 0, stream>>>(
      F[IN_SINGLE], F[IN_LNS_G], F[IN_LNS_B], tmp, tmpbf, CSD);
  launch_gemm(stream, tmpbf, CSD, 0, wsl, CSD, 0, F[IN_SL_B], nullptr, 0,
              s, CSD, 0, sbf, CSD, 0, NRES, CSD, CSD, 0, 1);

  for (int l = 0; l < NLAY; l++) {
    launch_gemm(stream, sbf, CSD, 0, wq,  192, 0, F[IN_Q_B],  nullptr, 0,
                qf, 192, 0, nullptr, 0, 0, NRES, 192, CSD, 0, 1);
    launch_gemm(stream, sbf, CSD, 0, wk,  192, 0, F[IN_K_B],  nullptr, 0,
                kf, 192, 0, nullptr, 0, 0, NRES, 192, CSD, 0, 1);
    launch_gemm(stream, sbf, CSD, 0, wv,  192, 0, F[IN_V_B],  nullptr, 0,
                nullptr, 0, 0, vbf, 192, 0, NRES, 192, CSD, 0, 1);
    launch_gemm(stream, sbf, CSD, 0, wqp, 144, 0, F[IN_QP_B], nullptr, 0,
                qpf, 144, 0, nullptr, 0, 0, NRES, 144, CSD, 0, 1);
    launch_gemm(stream, sbf, CSD, 0, wkp, 144, 0, F[IN_KP_B], nullptr, 0,
                kpf, 144, 0, nullptr, 0, 0, NRES, 144, CSD, 0, 1);
    launch_gemm(stream, sbf, CSD, 0, wvp, 288, 0, F[IN_VP_B], nullptr, 0,
                vpf, 288, 0, nullptr, 0, 0, NRES, 288, CSD, 0, 1);
    points_kernel<<<dim3(NRES), dim3(128), 0, stream>>>(
        qf, kf, qpf, kpf, vpf, Rc, tc, F[IN_GAMMA], Ub, Vb, vgbf, qn, kn);
    {
      dim3 g(NRES / 32, NRES / 64, HH);
      gemm_nt<<<g, dim3(256), 0, stream>>>(Ub, 32, (long long)NRES * 32,
                                           Vb, 32, (long long)NRES * 32,
                                           dotf, NRES, NN, NRES, NRES, 32);
    }
    softmax_kernel<<<dim3(HH * NRES), dim3(256), 0, stream>>>(
        dotf, biaszb, qn, kn, F[IN_GAMMA], abf);
    launch_gemm(stream, abf, NRES, NN, vbf, 192, 16, nullptr, nullptr, 0,
                of, 192, 16, nullptr, 0, 0, NRES, 16, NRES, 0, HH);
    launch_gemm(stream, abf, NRES, NN, vgbf, 288, 24, nullptr, nullptr, 0,
                opgf, 288, 24, nullptr, 0, 0, NRES, 24, NRES, 0, HH);
    opair_kernel<<<dim3(NRES), dim3(256), 0, stream>>>(abf, zbf, featsb);
    feats_kernel<<<dim3(NRES), dim3(192), 0, stream>>>(of, opgf, Rc, tc, featsb);
    launch_gemm(stream, featsb, FD, 0, wout, CSD, 0, F[IN_OUT_B], s, CSD,
                s, CSD, 0, nullptr, 0, 0, NRES, CSD, FD, 0, 1);
    ln_kernel<<<dim3(NRES), dim3(128), 0, stream>>>(s, F[IN_LN1_G], F[IN_LN1_B],
                                                    s, sbf, CSD);
    launch_gemm(stream, sbf, CSD, 0, wt0, CSD, 0, F[IN_T0_B], nullptr, 0,
                nullptr, 0, 0, h1b, CSD, 0, NRES, CSD, CSD, 1, 1);
    launch_gemm(stream, h1b, CSD, 0, wt1, CSD, 0, F[IN_T1_B], nullptr, 0,
                nullptr, 0, 0, h2b, CSD, 0, NRES, CSD, CSD, 1, 1);
    launch_gemm(stream, h2b, CSD, 0, wt2, CSD, 0, F[IN_T2_B], s, CSD,
                s, CSD, 0, nullptr, 0, 0, NRES, CSD, CSD, 0, 1);
    ln_kernel<<<dim3(NRES), dim3(128), 0, stream>>>(s, F[IN_LN2_G], F[IN_LN2_B],
                                                    s, sbf, CSD);
    launch_gemm(stream, sbf, CSD, 0, wbb, 6, 0, F[IN_BB_B], nullptr, 0,
                uf, 6, 0, nullptr, 0, 0, NRES, 6, CSD, 0, 1);
    backbone_kernel<<<dim3(12), dim3(64), 0, stream>>>(uf, Rc, tc);
    fape_kernel<<<dim3(NRES), dim3(256), 0, stream>>>(Rc, tc, F[IN_ROT], F[IN_TRANS],
                                                      &accs[ACC_FAPE0 + l]);
  }

  coords_kernel<<<dim3(NRES), dim3(32), 0, stream>>>(Rc, tc, F[IN_LOCAL], out);
  lddt_true_kernel<<<dim3(NRES), dim3(256), 0, stream>>>(tc, F[IN_TRANS], ltru, accs);
  launch_gemm(stream, sbf, CSD, 0, wl0, LCH, 0, F[IN_L0_B], nullptr, 0,
              nullptr, 0, 0, l1b, LCH, 0, NRES, LCH, CSD, 1, 1);
  launch_gemm(stream, l1b, LCH, 0, wl1, LCH, 0, F[IN_L1_B], nullptr, 0,
              nullptr, 0, 0, l2b, LCH, 0, NRES, LCH, LCH, 1, 1);
  launch_gemm(stream, l2b, LBINS == 50 ? LCH : LCH, 0, wl2, LBINS, 0, F[IN_L2_B], nullptr, 0,
              lgt, LBINS, 0, nullptr, 0, 0, NRES, LBINS, LCH, 0, 1);
  conf_kernel<<<dim3(NRES), dim3(64), 0, stream>>>(lgt, ltru, accs);
  finalize_kernel<<<dim3(1), dim3(32), 0, stream>>>(accs, out);
}
